// Attention_3934190044008
// MI455X (gfx1250) — matrix operand order repaired and checked by reading, not compiled, not run on hardware
//
#include <hip/hip_runtime.h>

typedef __attribute__((ext_vector_type(16))) _Float16 v16h;
typedef __attribute__((ext_vector_type(8)))  _Float16 v8h;
typedef __attribute__((ext_vector_type(8)))  float    v8f;
typedef __attribute__((ext_vector_type(4)))  float    v4f;

#define WMMA_F16(A, B, C) \
  __builtin_amdgcn_wmma_f32_16x16x32_f16(false, (A), false, (B), (short)0, (C), false, false)

// Problem constants
#define BD 2
#define SQ 2048
#define CQ 256
#define HD 512
#define NH 8
#define DH 64

// ---------------------------------------------------------------------------
// Tiny one-pass converters (~7 MB total, negligible).
// ---------------------------------------------------------------------------
__global__ void cvt_f16_kernel(const float* __restrict__ src,
                               _Float16* __restrict__ dst, int n)
{
  int i = blockIdx.x * 256 + threadIdx.x;
  if (i < n) dst[i] = (_Float16)src[i];
}

// src [R][C] f32 (C = 1<<logC) -> dst [C][R] f16, scaled.
__global__ void cvt_T_kernel(const float* __restrict__ src,
                             _Float16* __restrict__ dst, int logC, int R, float scale)
{
  int i = blockIdx.x * 256 + threadIdx.x;
  int C = 1 << logC;
  if (i < (R << logC)) {
    int r = i >> logC;
    int c = i & (C - 1);
    dst[(size_t)c * R + r] = (_Float16)(src[i] * scale);
  }
}

// ---------------------------------------------------------------------------
// Fused QKVG projections, all-f16 operands (pure b128 loads).
//   q -> f16 [BH, S, 64] (1/sqrt(64) folded into wqT)     k -> f16 [BH, S, 64]
//   v -> f16 [BH, 64, S] (transposed)                     g -> f32 [B*S, 512]
// ---------------------------------------------------------------------------
__global__ __launch_bounds__(128) void proj_kernel(
    const _Float16* __restrict__ xqh, const _Float16* __restrict__ xkh,
    const _Float16* __restrict__ wqT, const _Float16* __restrict__ wkT,
    const _Float16* __restrict__ wvT, const _Float16* __restrict__ wgT,
    const float* __restrict__ b_g,
    _Float16* __restrict__ qh, _Float16* __restrict__ kh,
    _Float16* __restrict__ vT, float* __restrict__ gbuf)
{
  const int wv   = threadIdx.x >> 5;
  const int lane = threadIdx.x & 31;
  const int lo   = lane & 15;
  const int hi   = lane >> 4;
  const int rt   = blockIdx.x;
  const int ct   = blockIdx.y * 4 + wv;
  const int pj   = blockIdx.z;

  const _Float16* x  = (pj == 0 || pj == 3) ? xqh : xkh;
  const _Float16* wT = (pj == 0) ? wqT : (pj == 1) ? wkT : (pj == 2) ? wvT : wgT;

  const int rowA = rt * 16 + lo;
  const int colg = ct * 16 + lo;
  const _Float16* ar = x  + (size_t)rowA * CQ;
  const _Float16* br = wT + (size_t)colg * CQ;

  v8f acc = {};
#pragma unroll
  for (int kk = 0; kk < CQ; kk += 32) {
    v16h a, bfr;
    ((v8h*)&a)[0]   = *(const v8h*)(ar + kk + 16 * hi);
    ((v8h*)&a)[1]   = *(const v8h*)(ar + kk + 16 * hi + 8);
    ((v8h*)&bfr)[0] = *(const v8h*)(br + kk + 16 * hi);
    ((v8h*)&bfr)[1] = *(const v8h*)(br + kk + 16 * hi + 8);
    acc = WMMA_F16(a, bfr, acc);
  }

  const int h = colg >> 6;
  const int d = colg & 63;
  if (pj == 3) {
    const float bgv = b_g[colg];
#pragma unroll
    for (int r = 0; r < 8; ++r) {
      const int m = rt * 16 + r + 8 * hi;
      gbuf[(size_t)m * HD + colg] = 1.0f / (1.0f + __expf(-(acc[r] + bgv)));
    }
  } else if (pj == 0) {
#pragma unroll
    for (int r = 0; r < 8; ++r) {
      const int m = rt * 16 + r + 8 * hi;
      qh[((size_t)((m >> 11) * NH + h) * SQ + (m & 2047)) * DH + d] = (_Float16)acc[r];
    }
  } else if (pj == 1) {
#pragma unroll
    for (int r = 0; r < 8; ++r) {
      const int m = rt * 16 + r + 8 * hi;
      kh[((size_t)((m >> 11) * NH + h) * SQ + (m & 2047)) * DH + d] = (_Float16)acc[r];
    }
  } else {
#pragma unroll
    for (int r = 0; r < 8; ++r) {
      const int m = rt * 16 + r + 8 * hi;
      vT[((size_t)((m >> 11) * NH + h) * DH + d) * SQ + (m & 2047)] = (_Float16)acc[r];
    }
  }
}

// ---------------------------------------------------------------------------
// Fused flash attention (transposed-tile formulation) with bias + gating.
// Computes S^T = K Q^T + bias^T and O^T = V^T P^T so that q lives in the
// lane index: per-lane softmax stats, vector bias/gate/P accesses.
// grid (16, 32) block 128 : wave = 16 q-rows, kv tiles of 64.
// ---------------------------------------------------------------------------
__global__ __launch_bounds__(128) void attn_kernel(
    const _Float16* __restrict__ qh, const _Float16* __restrict__ kh,
    const _Float16* __restrict__ vT, const float* __restrict__ bias,
    const float* __restrict__ gbuf, _Float16* __restrict__ obuf)
{
  __shared__ __align__(16) _Float16 lds_p[4][16 * 64];   // wave-private P [q][kv]

  const int bh   = blockIdx.x;                 // b*8+h
  const int qt   = blockIdx.y;
  const int wv   = threadIdx.x >> 5;
  const int lane = threadIdx.x & 31;
  const int lo   = lane & 15;
  const int hi   = lane >> 4;
  const int q0   = qt * 64 + wv * 16;
  const int b    = bh >> 3;
  const int h    = bh & 7;

  // Q^T B-fragments (lane = q-column), loaded once: contiguous 16-half runs.
  v16h qb[2];
  {
    const _Float16* qrow = qh + ((size_t)bh * SQ + q0 + lo) * DH;
#pragma unroll
    for (int c = 0; c < 2; ++c) {
      ((v8h*)&qb[c])[0] = *(const v8h*)(qrow + 32 * c + 16 * hi);
      ((v8h*)&qb[c])[1] = *(const v8h*)(qrow + 32 * c + 16 * hi + 8);
    }
  }

  const float* bias_row = bias + ((size_t)bh * SQ + q0 + lo) * SQ;          // this lane's q
  const float* bias_pf  = bias + ((size_t)bh * SQ + q0 + (lane >> 1)) * SQ + (lane & 1) * 32;

  v8f acc[4] = {};                             // O^T: 4 d-blocks (rows d, col q)
  float mrun = -1e30f, lrun = 0.0f;            // per-lane (= per-q) stats

  for (int kt = 0; kt < SQ / 64; ++kt) {
    const int k0 = kt * 64;

    // Prefetch bias two k-tiles ahead: 16 rows x 256B = 32 x 128B lines.
    if (k0 + 128 < SQ) __builtin_prefetch(bias_pf + k0 + 128, 0, 0);

    // ---- S^T = K Q^T + bias^T : 4 tiles of 16 kv rows --------------------
    v8f s[4];
#pragma unroll
    for (int nb = 0; nb < 4; ++nb) {
      v8f sv;  // C operand: bias[q=lo][kv = k0+nb*16+8*hi+r], contiguous in r
      const float* bp = bias_row + k0 + nb * 16 + 8 * hi;
      ((v4f*)&sv)[0] = __builtin_nontemporal_load((const v4f*)bp);
      ((v4f*)&sv)[1] = __builtin_nontemporal_load((const v4f*)(bp + 4));
#pragma unroll
      for (int c = 0; c < 2; ++c) {
        v16h ka;  // A operand: K rows (lane = kv row), contiguous d runs
        const _Float16* kr = kh + ((size_t)bh * SQ + k0 + nb * 16 + lo) * DH + 32 * c;
        ((v8h*)&ka)[0] = *(const v8h*)(kr + 16 * hi);
        ((v8h*)&ka)[1] = *(const v8h*)(kr + 16 * hi + 8);
        sv = WMMA_F16(ka, qb[c], sv);
      }
      s[nb] = sv;
    }

    // ---- online softmax: stats are per-lane; lane pair (l, l+16) holds the
    // two kv halves of the same q -> a single shfl_xor(16) merges them.
    float rm = -1e30f;
#pragma unroll
    for (int nb = 0; nb < 4; ++nb)
#pragma unroll
      for (int r = 0; r < 8; ++r) rm = fmaxf(rm, s[nb][r]);
    rm = fmaxf(rm, __shfl_xor(rm, 16, 32));
    const float mnew  = fmaxf(mrun, rm);
    const float scale = __expf(mrun - mnew);
#pragma unroll
    for (int db = 0; db < 4; ++db)
#pragma unroll
      for (int r = 0; r < 8; ++r) acc[db][r] *= scale;
    lrun *= scale;
    float ps = 0.0f;
#pragma unroll
    for (int nb = 0; nb < 4; ++nb)
#pragma unroll
      for (int r = 0; r < 8; ++r) {
        const float p = __expf(s[nb][r] - mnew);
        s[nb][r] = p;
        ps += p;
      }
    ps += __shfl_xor(ps, 16, 32);
    lrun += ps;
    mrun = mnew;

    // ---- P -> LDS row-major [q][kv]: packed b128 stores ------------------
    _Float16* pl = lds_p[wv];
#pragma unroll
    for (int nb = 0; nb < 4; ++nb) {
      v8h ph;
#pragma unroll
      for (int r = 0; r < 8; ++r) ph[r] = (_Float16)s[nb][r];
      *(v8h*)(pl + lo * 64 + nb * 16 + 8 * hi) = ph;
    }
    // Same-wave LDS ops are in-order; one DScnt wait replaces s_barrier.
    asm volatile("s_wait_dscnt 0x0" ::: "memory");

    // P^T B-fragments (lane = q-column): contiguous 16-half kv runs.
    v16h pb[2];
#pragma unroll
    for (int c = 0; c < 2; ++c) {
      const _Float16* pr = pl + lo * 64 + 32 * c + 16 * hi;
      ((v8h*)&pb[c])[0] = *(const v8h*)(pr);
      ((v8h*)&pb[c])[1] = *(const v8h*)(pr + 8);
    }

    // ---- O^T += V^T P^T (A = V^T rows, contiguous kv runs) ---------------
#pragma unroll
    for (int db = 0; db < 4; ++db) {
#pragma unroll
      for (int c = 0; c < 2; ++c) {
        v16h va;
        const _Float16* vr = vT + ((size_t)bh * DH + db * 16 + lo) * SQ + k0 + 32 * c;
        ((v8h*)&va)[0] = *(const v8h*)(vr + 16 * hi);
        ((v8h*)&va)[1] = *(const v8h*)(vr + 16 * hi + 8);
        acc[db] = WMMA_F16(va, pb[c], acc[db]);
      }
    }
  }

  // ---- epilogue: O^T -> normalize, gate (vector loads), store f16 --------
  const int qrow = q0 + lo;
  const float inv = 1.0f / lrun;
  const float*  gp = gbuf + ((size_t)b * SQ + qrow) * HD + h * DH;
  _Float16*     op = obuf + ((size_t)b * SQ + qrow) * HD + h * DH;
#pragma unroll
  for (int db = 0; db < 4; ++db) {
    const int doff = db * 16 + 8 * hi;         // lane's d run: 8 consecutive
    v4f g0 = *(const v4f*)(gp + doff);
    v4f g1 = *(const v4f*)(gp + doff + 4);
    v8h ph;
#pragma unroll
    for (int j = 0; j < 4; ++j) ph[j]     = (_Float16)(acc[db][j]     * inv * g0[j]);
#pragma unroll
    for (int j = 0; j < 4; ++j) ph[4 + j] = (_Float16)(acc[db][4 + j] * inv * g1[j]);
    *(v8h*)(op + doff) = ph;
  }
}

// ---------------------------------------------------------------------------
// out = obuf[4096,512] @ w_o + b_o -> f32 [B*S, 256]
// ---------------------------------------------------------------------------
__global__ __launch_bounds__(128) void outproj_kernel(
    const _Float16* __restrict__ obuf, const _Float16* __restrict__ woT,
    const float* __restrict__ b_o, float* __restrict__ out)
{
  const int wv   = threadIdx.x >> 5;
  const int lane = threadIdx.x & 31;
  const int lo   = lane & 15;
  const int hi   = lane >> 4;
  const int rt   = blockIdx.x;
  const int ct   = blockIdx.y * 4 + wv;

  const _Float16* ar = obuf + (size_t)(rt * 16 + lo) * HD;
  const _Float16* br = woT  + (size_t)(ct * 16 + lo) * HD;

  v8f acc = {};
#pragma unroll
  for (int kk = 0; kk < HD; kk += 32) {
    v16h a, bfr;
    ((v8h*)&a)[0]   = *(const v8h*)(ar + kk + 16 * hi);
    ((v8h*)&a)[1]   = *(const v8h*)(ar + kk + 16 * hi + 8);
    ((v8h*)&bfr)[0] = *(const v8h*)(br + kk + 16 * hi);
    ((v8h*)&bfr)[1] = *(const v8h*)(br + kk + 16 * hi + 8);
    acc = WMMA_F16(a, bfr, acc);
  }

  const float bo = b_o[ct * 16 + lo];
#pragma unroll
  for (int r = 0; r < 8; ++r)
    out[(size_t)(rt * 16 + r + 8 * hi) * CQ + ct * 16 + lo] = acc[r] + bo;
}

// ---------------------------------------------------------------------------
extern "C" void kernel_launch(void* const* d_in, const int* in_sizes, int n_in,
                              void* d_out, int out_size, void* d_ws, size_t ws_size,
                              hipStream_t stream)
{
  (void)in_sizes; (void)n_in; (void)out_size; (void)ws_size;
  const float* q_x  = (const float*)d_in[0];
  const float* kv_x = (const float*)d_in[1];
  const float* bias = (const float*)d_in[2];
  const float* w_q  = (const float*)d_in[3];
  const float* w_k  = (const float*)d_in[4];
  const float* w_v  = (const float*)d_in[5];
  const float* w_g  = (const float*)d_in[6];
  const float* b_g  = (const float*)d_in[7];
  const float* w_o  = (const float*)d_in[8];
  const float* b_o  = (const float*)d_in[9];
  float* out = (float*)d_out;

  char* ws = (char*)d_ws;
  size_t off = 0;
  auto take = [&](size_t bytes) { char* p = ws + off; off += (bytes + 255) & ~(size_t)255; return p; };
  _Float16* qh   = (_Float16*)take((size_t)BD * NH * SQ * DH * 2);
  _Float16* kh   = (_Float16*)take((size_t)BD * NH * SQ * DH * 2);
  _Float16* vT   = (_Float16*)take((size_t)BD * NH * DH * SQ * 2);
  _Float16* obuf = (_Float16*)take((size_t)BD * SQ * HD * 2);
  float*    gbuf = (float*)   take((size_t)BD * SQ * HD * 4);
  _Float16* xqh  = (_Float16*)take((size_t)BD * SQ * CQ * 2);
  _Float16* xkh  = (_Float16*)take((size_t)BD * SQ * CQ * 2);
  _Float16* wqT  = (_Float16*)take((size_t)CQ * HD * 2);
  _Float16* wkT  = (_Float16*)take((size_t)CQ * HD * 2);
  _Float16* wvT  = (_Float16*)take((size_t)CQ * HD * 2);
  _Float16* wgT  = (_Float16*)take((size_t)CQ * HD * 2);
  _Float16* woT  = (_Float16*)take((size_t)HD * CQ * 2);

  const int nX = BD * SQ * CQ;
  cvt_f16_kernel<<<(nX + 255) / 256, 256, 0, stream>>>(q_x, xqh, nX);
  cvt_f16_kernel<<<(nX + 255) / 256, 256, 0, stream>>>(kv_x, xkh, nX);
  const int nW = CQ * HD;
  cvt_T_kernel<<<(nW + 255) / 256, 256, 0, stream>>>(w_q, wqT, 9, CQ, 0.125f);
  cvt_T_kernel<<<(nW + 255) / 256, 256, 0, stream>>>(w_k, wkT, 9, CQ, 1.0f);
  cvt_T_kernel<<<(nW + 255) / 256, 256, 0, stream>>>(w_v, wvT, 9, CQ, 1.0f);
  cvt_T_kernel<<<(nW + 255) / 256, 256, 0, stream>>>(w_g, wgT, 9, CQ, 1.0f);
  cvt_T_kernel<<<(nW + 255) / 256, 256, 0, stream>>>(w_o, woT, 8, HD, 1.0f);

  proj_kernel<<<dim3(256, 8, 4), 128, 0, stream>>>(xqh, xkh, wqT, wkT, wvT, wgT,
                                                   b_g, qh, kh, vT, gbuf);
  attn_kernel<<<dim3(BD * NH, SQ / 64), 128, 0, stream>>>(qh, kh, vT, bias, gbuf, obuf);
  outproj_kernel<<<dim3(256, 4), 128, 0, stream>>>(obuf, woT, b_o, out);
}
